// HebbianMemory_25804163514992
// MI455X (gfx1250) — matrix operand order repaired and checked by reading, not compiled, not run on hardware
//
#include <hip/hip_runtime.h>
#include <hip/hip_bf16.h>
#include <math.h>

typedef __attribute__((ext_vector_type(16))) _Float16 v16h;
typedef __attribute__((ext_vector_type(8)))  float    v8f;
typedef __attribute__((ext_vector_type(4)))  int      v4i;

typedef __attribute__((address_space(1))) v4i* as1_v4i_p;
typedef __attribute__((address_space(3))) v4i* as3_v4i_p;

#define B_   32
#define N_   4096
#define DK   256
#define DV   256
#define TILE_M 64
#define TOPK 16

// ---------------- workspace layout ----------------
// [0,        8.0 MB) : Ht   f32  [B][DV][DK]   (H transposed, un-normalized, sparse gather source)
// [8.0 MB, 12.2 MB) : HnF  f16  [B][16 nt][8 kt][32 lane][16 h]  (normalized H, WMMA-B frag order)
#define HT_ELEMS   ((size_t)B_ * DV * DK)
#define HNF_OFF    (HT_ELEMS * sizeof(float))
#define HNF_B_BYTES ((size_t)16 * 8 * 512 * sizeof(_Float16))   // 128 KB per batch

// ---------------- LDS layout (dynamic, 224 KB) ----------------
#define LDS_AF_BYTES   (4 * 8 * 32 * 16 * sizeof(_Float16))     //  32 KB
#define LDS_BF_BYTES   HNF_B_BYTES                              // 128 KB
#define LDS_SIMS_BYTES ((size_t)TILE_M * DV * sizeof(float))    //  64 KB
#define LDS_TOTAL      (LDS_AF_BYTES + LDS_BF_BYTES + LDS_SIMS_BYTES)

#if defined(__has_builtin)
#if __has_builtin(__builtin_amdgcn_global_load_async_to_lds_b128) && \
    __has_builtin(__builtin_amdgcn_s_wait_asynccnt)
#define USE_ASYNC_LDS 1
#endif
#if __has_builtin(__builtin_amdgcn_sched_group_barrier)
#define SCHED_GROUP(mask, size, id) __builtin_amdgcn_sched_group_barrier((mask), (size), (id))
#endif
#endif
#ifndef SCHED_GROUP
#define SCHED_GROUP(mask, size, id)
#endif

// fast reciprocal / rsqrt (normalization + softmax scaling tolerate 1-ulp rcp)
__device__ __forceinline__ float fast_rcp(float x)  { return __builtin_amdgcn_rcpf(x); }
__device__ __forceinline__ float fast_rsq(float x)  { return __builtin_amdgcn_rsqf(x); }

// =====================================================================
// Prep: one wave per H column j.  Normalize H[b][:][j] over d_k, emit:
//   Ht[b][j][d] (f32, row-contiguous)  and  HnF in WMMA B-fragment layout.
// B fragment order used here (contiguous k, same as A): lane = (k<16 ? n : 16+n), half = k&15.
// =====================================================================
__global__ void hebbian_prep_kernel(const float* __restrict__ H,
                                    float* __restrict__ Ht,
                                    _Float16* __restrict__ HnF) {
    const int tid  = threadIdx.x;
    const int w    = tid >> 5;
    const int lane = tid & 31;
    const int gw   = blockIdx.x * 8 + w;          // global wave id = column id
    const int b    = gw >> 8;
    const int j    = gw & 255;

    float v[8];
    float ss = 0.f;
#pragma unroll
    for (int i = 0; i < 8; ++i) {
        const int d = lane + 32 * i;
        v[i] = H[((size_t)b * DK + d) * DV + j];
        ss += v[i] * v[i];
    }
#pragma unroll
    for (int off = 16; off > 0; off >>= 1) ss += __shfl_xor(ss, off, 32);
    const float rn = fast_rsq(fmaxf(ss, 1e-16f));   // 1/max(||x||, eps)

    const int nt = j >> 4, nn = j & 15;
#pragma unroll
    for (int i = 0; i < 8; ++i) {
        const int d = lane + 32 * i;
        Ht[((size_t)b * DV + j) * DK + d] = v[i];
        const int kt = d >> 5, kk = d & 31;
        const int fl = (kk < 16 ? nn : 16 + nn);
        const int fh = kk & 15;
        HnF[(((size_t)b * 16 + nt) * 8 + kt) * 512 + fl * 16 + fh] = (_Float16)(v[i] * rn);
    }
}

// =====================================================================
// Fused main kernel: 256 threads (8 wave32), 64 rows of one batch.
// =====================================================================
__global__ void __launch_bounds__(256)
hebbian_main_kernel(const float* __restrict__ C,
                    const _Float16* __restrict__ HnF,
                    const float* __restrict__ Ht,
                    const float* __restrict__ temp,
                    float* __restrict__ out) {
    extern __shared__ __align__(16) char smem[];
    _Float16* aF   = (_Float16*)smem;
    _Float16* bF   = (_Float16*)(smem + LDS_AF_BYTES);
    float*    sims = (float*)(smem + LDS_AF_BYTES + LDS_BF_BYTES);

    const int tid  = threadIdx.x;
    const int w    = tid >> 5;
    const int lane = tid & 31;
    const int b    = blockIdx.x >> 6;
    const int n0   = (blockIdx.x & 63) * TILE_M;

    const float t     = fmaxf(temp[0], 0.1f);
    const float inv_t = fast_rcp(t + 1e-8f);

    // ---------------- Phase 0: async-copy this batch's B fragments (128 KB) to LDS.
    // Overlaps with Phase 1 C normalization; completion enforced before GEMM.
    {
        char* gsrc = const_cast<char*>((const char*)(HnF + (size_t)b * 16 * 8 * 512));
        char* ldst = (char*)bF;
#pragma unroll
        for (int i = 0; i < 32; ++i) {
            const size_t off = ((size_t)tid + 256 * i) * 16;
#ifdef USE_ASYNC_LDS
            __builtin_amdgcn_global_load_async_to_lds_b128(
                (as1_v4i_p)(gsrc + off),
                (as3_v4i_p)(ldst + off), 0, 0);
#else
            *(v4i*)(ldst + off) = *(const v4i*)(gsrc + off);
#endif
        }
    }

    // ---------------- Phase 1: row-normalize C tile -> f16 A-fragments in LDS.
    // For this A layout, lane's 8 k-values (k = lane*8 .. lane*8+7) occupy 8
    // consecutive halves at a 16B-aligned LDS address -> one ds_store_b128.
    for (int rr = 0; rr < 8; ++rr) {
        const int row = w * 8 + rr;
        const float* cp = C + ((size_t)b * N_ + n0 + row) * DK + lane * 8;
        float cv[8];
        float ss = 0.f;
#pragma unroll
        for (int i = 0; i < 8; ++i) { cv[i] = cp[i]; ss += cv[i] * cv[i]; }
#pragma unroll
        for (int off = 16; off > 0; off >>= 1) ss += __shfl_xor(ss, off, 32);
        const float rn = fast_rsq(fmaxf(ss, 1e-16f));

        const int m  = row & 15, mt = row >> 4;
        const int k0 = lane * 8;
        const int kt = k0 >> 5;                 // all 8 values share one kt
        const int lh = (k0 >> 4) & 1;           // lane-half select
        const int hb = ((k0 >> 3) & 1) * 8;     // half-base within 16
        const int fl = m + lh * 16;

        _Float16 tmp[8];
#pragma unroll
        for (int i = 0; i < 8; ++i) tmp[i] = (_Float16)(cv[i] * rn);
        *(v4i*)&aF[(((size_t)mt * 8 + kt) * 32 + fl) * 16 + hb] = *(const v4i*)tmp;
    }

#ifdef USE_ASYNC_LDS
    __builtin_amdgcn_s_wait_asynccnt(0);
#endif
    __syncthreads();

    // ---------------- Phase 2: WMMA GEMM  sims(64x256) = Cn(64x256) x Hn(256x256)
    const int mt     = w >> 1;
    const int ntbase = (w & 1) * 8;

    v8f acc[8];
#pragma unroll
    for (int i = 0; i < 8; ++i) acc[i] = (v8f)0.0f;

#pragma unroll
    for (int kt = 0; kt < 8; ++kt) {
        const v16h a = *(const v16h*)&aF[(((size_t)mt * 8 + kt) * 32 + lane) * 16];
        v16h bm[8];
#pragma unroll
        for (int q = 0; q < 8; ++q)
            bm[q] = *(const v16h*)&bF[(((size_t)(ntbase + q) * 8 + kt) * 512) + lane * 16];
#pragma unroll
        for (int q = 0; q < 8; ++q)
            acc[q] = __builtin_amdgcn_wmma_f32_16x16x32_f16(
                false, a, false, bm[q], (short)0, acc[q], false, false);
        // pipeline: issue all 18 DS reads (A:2 + B:16) of this iteration,
        // then the 8 WMMAs -> staggered s_wait_dscnt instead of dscnt==0 stalls
        SCHED_GROUP(0x100, 18, 0);   // DS reads
        SCHED_GROUP(0x008, 8, 0);    // WMMA
    }

    // scatter accumulators to sims LDS.
    // C/D layout: VGPR r, lanes0-15: M=r,N=lane ; lanes16-31: M=8+r,N=lane-16
    const int nlo   = lane & 15;
    const int mhalf = (lane >> 4) * 8;
#pragma unroll
    for (int q = 0; q < 8; ++q) {
        const int col = (ntbase + q) * 16 + nlo;
#pragma unroll
        for (int r = 0; r < 8; ++r) {
            const int row = mt * 16 + mhalf + r;
            sims[row * DV + col] = acc[q][r];
        }
    }
    __syncthreads();

    // ---------------- Phase 3: softmax + top-16 + renorm + sparse gather
    const float* HtB = Ht + (size_t)b * DV * DK;
    for (int rr = 0; rr < 8; ++rr) {
        const int row = w * 8 + rr;

        float vals[8];
#pragma unroll
        for (int i = 0; i < 8; ++i) vals[i] = sims[row * DV + lane + 32 * i];

        float mx = vals[0];
#pragma unroll
        for (int i = 1; i < 8; ++i) mx = fmaxf(mx, vals[i]);
#pragma unroll
        for (int off = 16; off > 0; off >>= 1) mx = fmaxf(mx, __shfl_xor(mx, off, 32));

        float sfull = 0.f;
#pragma unroll
        for (int i = 0; i < 8; ++i) {
            vals[i] = __expf((vals[i] - mx) * inv_t);
            sfull += vals[i];
        }
#pragma unroll
        for (int off = 16; off > 0; off >>= 1) sfull += __shfl_xor(sfull, off, 32);

        // iterative top-16 (wave-wide argmax, then mask winner)
        float selv[TOPK];
        int   selc[TOPK];
        float work[8];
#pragma unroll
        for (int i = 0; i < 8; ++i) work[i] = vals[i];
#pragma unroll
        for (int rsel = 0; rsel < TOPK; ++rsel) {
            float bv = work[0];
            int   bc = lane;
#pragma unroll
            for (int i = 1; i < 8; ++i) {
                const int c = lane + 32 * i;
                if (work[i] > bv) { bv = work[i]; bc = c; }
            }
#pragma unroll
            for (int off = 16; off > 0; off >>= 1) {
                const float ov = __shfl_xor(bv, off, 32);
                const int   oc = __shfl_xor(bc, off, 32);
                if (ov > bv || (ov == bv && oc < bc)) { bv = ov; bc = oc; }
            }
            selv[rsel] = bv;
            selc[rsel] = bc;
            if ((bc & 31) == lane) work[bc >> 5] = -3.4e38f;
        }

        float ssel = 0.f;
#pragma unroll
        for (int rsel = 0; rsel < TOPK; ++rsel) ssel += selv[rsel];
        const float denom = ssel + 1e-8f * sfull;
        const float rden  = fast_rcp(denom);

        // sparse gather, vectorized: lane owns d = lane*8 .. lane*8+7
        float o[8];
#pragma unroll
        for (int i = 0; i < 8; ++i) o[i] = 0.f;
#pragma unroll
        for (int rsel = 0; rsel < TOPK; ++rsel) {
            const float wr = selv[rsel] * rden;
            const float4* hp4 = (const float4*)(HtB + (size_t)selc[rsel] * DK) + lane * 2;
            const float4 p0 = hp4[0];
            const float4 p1 = hp4[1];
            o[0] += wr * p0.x; o[1] += wr * p0.y; o[2] += wr * p0.z; o[3] += wr * p0.w;
            o[4] += wr * p1.x; o[5] += wr * p1.y; o[6] += wr * p1.z; o[7] += wr * p1.w;
        }
        // issue all 32 gather loads first, then the FMA block (max MLP)
        SCHED_GROUP(0x020, 32, 0);   // VMEM reads
        SCHED_GROUP(0x002, 256, 0);  // VALU

        float* op = out + ((size_t)b * N_ + n0 + row) * DK + lane * 8;
        float4 s0 = make_float4(o[0], o[1], o[2], o[3]);
        float4 s1 = make_float4(o[4], o[5], o[6], o[7]);
        *(float4*)(op)     = s0;
        *(float4*)(op + 4) = s1;
    }
}

extern "C" void kernel_launch(void* const* d_in, const int* in_sizes, int n_in,
                              void* d_out, int out_size, void* d_ws, size_t ws_size,
                              hipStream_t stream) {
    const float* C    = (const float*)d_in[0];
    const float* H    = (const float*)d_in[1];
    const float* temp = (const float*)d_in[2];
    float* out = (float*)d_out;

    float*    Ht  = (float*)d_ws;
    _Float16* HnF = (_Float16*)((char*)d_ws + HNF_OFF);

    hipLaunchKernelGGL(hebbian_prep_kernel,
                       dim3((B_ * DV) / 8), dim3(256), 0, stream, H, Ht, HnF);

    hipLaunchKernelGGL(hebbian_main_kernel,
                       dim3(B_ * (N_ / TILE_M)), dim3(256), LDS_TOTAL, stream,
                       C, HnF, Ht, temp, out);
}
